// GNNLayer_55834574848370
// MI455X (gfx1250) — matrix operand order repaired and checked by reading, not compiled, not run on hardware
//
#include <hip/hip_runtime.h>
#include <hip/hip_bf16.h>
#include <math.h>

// GNN GAT layer for MI455X (gfx1250, wave32, WMMA).
// Pipeline per head: f16 WMMA GEMM1 (ReLU) -> f16 WMMA GEMM2 -> edge scores
// (+global max) -> softmax-weighted scatter (native f32 atomics) -> divide.

typedef _Float16 f16;
typedef _Float16 v16h __attribute__((ext_vector_type(16)));
typedef _Float16 v8h  __attribute__((ext_vector_type(8)));
typedef float    v8f  __attribute__((ext_vector_type(8)));

#define N_NODES 50000
#define N_EDGES 800000
#define IN_DIM  256
#define HID_DIM 256
#define OUT_DIM 64
#define N_HEADS 4

// ---- ordered-uint encoding of float for atomicMax --------------------------
__device__ __forceinline__ unsigned enc_f32(float f) {
  unsigned u = __float_as_uint(f);
  return (u & 0x80000000u) ? ~u : (u | 0x80000000u);
}
__device__ __forceinline__ float dec_f32(unsigned u) {
  return __uint_as_float((u & 0x80000000u) ? (u ^ 0x80000000u) : ~u);
}
__device__ __forceinline__ void atomAddF(float* p, float v) {
  // lowers to global_atomic_add_f32 (no-return form), relaxed, device scope
  __hip_atomic_fetch_add(p, v, __ATOMIC_RELAXED, __HIP_MEMORY_SCOPE_AGENT);
}

// ---- init: zero accumulators, set max = -inf -------------------------------
__global__ void k_init(float* __restrict__ out, float* __restrict__ rowsum,
                       unsigned* __restrict__ maxu) {
  size_t stride = (size_t)gridDim.x * blockDim.x;
  size_t i = (size_t)blockIdx.x * blockDim.x + threadIdx.x;
  for (size_t j = i; j < (size_t)N_NODES * 256; j += stride) out[j] = 0.0f;
  for (size_t j = i; j < (size_t)N_HEADS * N_NODES; j += stride) rowsum[j] = 0.0f;
  if (i < N_HEADS) maxu[i] = 0x007FFFFFu;  // enc(-inf)
}

// ---- fp32 -> f16 conversions ----------------------------------------------
__global__ void k_cvt_x(const float* __restrict__ x, f16* __restrict__ xh) {
  size_t stride = (size_t)gridDim.x * blockDim.x;
  for (size_t i = (size_t)blockIdx.x * blockDim.x + threadIdx.x;
       i < (size_t)N_NODES * IN_DIM; i += stride)
    xh[i] = (f16)x[i];
}
// W1[h][k][j] -> w1t[h][j][k]  (f16, transposed so WMMA B loads are contiguous)
__global__ void k_cvt_w1(const float* __restrict__ W1, f16* __restrict__ w1t) {
  int i = blockIdx.x * blockDim.x + threadIdx.x;   // 4*256*256 = 262144
  if (i >= N_HEADS * HID_DIM * IN_DIM) return;
  int h = i >> 16, rem = i & 65535, j = rem >> 8, k = rem & 255;
  w1t[i] = (f16)W1[(size_t)h * 65536 + (size_t)k * 256 + j];
}
// W2[h][k][o] -> w2t[h][o][k]
__global__ void k_cvt_w2(const float* __restrict__ W2, f16* __restrict__ w2t) {
  int i = blockIdx.x * blockDim.x + threadIdx.x;   // 4*64*256 = 65536
  if (i >= N_HEADS * OUT_DIM * HID_DIM) return;
  int h = i >> 14, rem = i & 16383, o = rem >> 8, k = rem & 255;
  w2t[i] = (f16)W2[(size_t)h * 16384 + (size_t)k * 64 + o];
}

// ---- WMMA fragment loaders (per documented CDNA5 wave32 layouts) -----------
// A (16x32 f16): lane = hi*16 + m; half t<8 -> K = hi*8 + t ; t>=8 -> K = 16 + hi*8 + (t-8)
__device__ __forceinline__ v16h load_a(const f16* rowBase /* = &A[m][hi*8] */, int k0) {
  v8h lo = *(const v8h*)(rowBase + k0);
  v8h hi8 = *(const v8h*)(rowBase + k0 + 16);
  return __builtin_shufflevector(lo, hi8, 0,1,2,3,4,5,6,7,8,9,10,11,12,13,14,15);
}
// B (32x16 f16) from transposed weights Wt[n][k]: lane = hi*16 + n; half t<8 -> K = hi*8 + t ; t>=8 -> K = 16 + hi*8 + (t-8)
__device__ __forceinline__ v16h load_b(const f16* colBase /* = &Wt[n][hi*8] */, int k0) {
  v8h lo = *(const v8h*)(colBase + k0);
  v8h hi8 = *(const v8h*)(colBase + k0 + 16);
  return __builtin_shufflevector(lo, hi8, 0,1,2,3,4,5,6,7,8,9,10,11,12,13,14,15);
}

// ---- GEMM1: hid[n][j] = relu(x @ W1h + b1h)  (f16 out) ---------------------
// grid (3125, 4), block 128 (4 waves); wave w covers 16 cols, block 16x64 tile
__global__ __launch_bounds__(128)
void k_gemm1(const f16* __restrict__ xh, const f16* __restrict__ w1t,
             const float* __restrict__ b1, f16* __restrict__ hid, int h) {
  int wave = threadIdx.x >> 5;
  int lane = threadIdx.x & 31;
  int r = lane & 15, hi = lane >> 4;
  int m0 = blockIdx.x * 16;
  int n0 = blockIdx.y * 64 + wave * 16;

  const f16* arow = xh + (size_t)(m0 + r) * IN_DIM + hi * 8;
  const f16* brow = w1t + ((size_t)h * HID_DIM + n0 + r) * IN_DIM + hi * 8;

  v8f acc = {};
#pragma unroll
  for (int k0 = 0; k0 < IN_DIM; k0 += 32) {
    if (k0 + 32 < IN_DIM) __builtin_prefetch(brow + k0 + 32, 0, 1);
    v16h a = load_a(arow, k0);
    v16h b = load_b(brow, k0);
    acc = __builtin_amdgcn_wmma_f32_16x16x32_f16(false, a, false, b,
                                                 (short)0, acc, false, false);
  }
  float bias = b1[h * HID_DIM + n0 + r];
#pragma unroll
  for (int v = 0; v < 8; ++v) {
    float val = fmaxf(acc[v] + bias, 0.0f);
    hid[(size_t)(m0 + v + hi * 8) * HID_DIM + n0 + r] = (f16)val;
  }
}

// ---- GEMM2: feats[n][o] = hid @ W2h + b2h  (f32 out) -----------------------
// grid (3125), block 128; wave w covers cols w*16..w*16+15 (OUT_DIM=64)
__global__ __launch_bounds__(128)
void k_gemm2(const f16* __restrict__ hid, const f16* __restrict__ w2t,
             const float* __restrict__ b2, float* __restrict__ feats, int h) {
  int wave = threadIdx.x >> 5;
  int lane = threadIdx.x & 31;
  int r = lane & 15, hi = lane >> 4;
  int m0 = blockIdx.x * 16;
  int n0 = wave * 16;

  const f16* arow = hid + (size_t)(m0 + r) * HID_DIM + hi * 8;
  const f16* brow = w2t + ((size_t)h * OUT_DIM + n0 + r) * HID_DIM + hi * 8;

  v8f acc = {};
#pragma unroll
  for (int k0 = 0; k0 < HID_DIM; k0 += 32) {
    if (k0 + 32 < HID_DIM) __builtin_prefetch(brow + k0 + 32, 0, 1);
    v16h a = load_a(arow, k0);
    v16h b = load_b(brow, k0);
    acc = __builtin_amdgcn_wmma_f32_16x16x32_f16(false, a, false, b,
                                                 (short)0, acc, false, false);
  }
  float bias = b2[h * OUT_DIM + n0 + r];
#pragma unroll
  for (int v = 0; v < 8; ++v)
    feats[(size_t)(m0 + v + hi * 8) * OUT_DIM + n0 + r] = acc[v] + bias;
}

// ---- edge scores + global max ----------------------------------------------
__global__ __launch_bounds__(256)
void k_score(const float* __restrict__ feats, const int* __restrict__ idx,
             const float* __restrict__ elem, const float* __restrict__ aw,
             const float* __restrict__ ab, float* __restrict__ score,
             unsigned* __restrict__ maxu, int h) {
  __shared__ float smax[256];
  int e = blockIdx.x * blockDim.x + threadIdx.x;
  float s = -INFINITY;
  if (e < N_EDGES) {
    int src = idx[e];
    int dst = idx[N_EDGES + e];
    const float* fs = feats + (size_t)src * OUT_DIM;
    const float* fd = feats + (size_t)dst * OUT_DIM;
    const float* w = aw + h * (2 * OUT_DIM + 1);
    float acc = w[2 * OUT_DIM] * elem[e] + ab[h];
#pragma unroll 8
    for (int o = 0; o < OUT_DIM; ++o) acc += w[o] * fs[o] + w[OUT_DIM + o] * fd[o];
    score[e] = acc;
    s = acc;
  }
  smax[threadIdx.x] = s;
  __syncthreads();
  for (int st = 128; st > 0; st >>= 1) {
    if (threadIdx.x < st)
      smax[threadIdx.x] = fmaxf(smax[threadIdx.x], smax[threadIdx.x + st]);
    __syncthreads();
  }
  if (threadIdx.x == 0) atomicMax(maxu + h, enc_f32(smax[0]));
}

// ---- softmax-weighted scatter: one wave32 per edge -------------------------
__global__ __launch_bounds__(256)
void k_scatter(const float* __restrict__ feats, const int* __restrict__ idx,
               const float* __restrict__ score, const unsigned* __restrict__ maxu,
               float* __restrict__ out, float* __restrict__ rowsum, int h) {
  int gid = blockIdx.x * blockDim.x + threadIdx.x;
  int e = gid >> 5;
  int lane = gid & 31;
  if (e >= N_EDGES) return;
  float mx = dec_f32(maxu[h]);
  float w = expf(score[e] - mx);
  int src = idx[e];
  int dst = idx[N_EDGES + e];
  const float* fd = feats + (size_t)dst * OUT_DIM;
  float* o0 = out + (size_t)src * (N_HEADS * OUT_DIM) + h * OUT_DIM;
  atomAddF(o0 + lane,      w * fd[lane]);
  atomAddF(o0 + lane + 32, w * fd[lane + 32]);
  if (lane == 0) atomAddF(rowsum + (size_t)h * N_NODES + src, w);
}

// ---- final divide ----------------------------------------------------------
__global__ void k_div(float* __restrict__ out, const float* __restrict__ rowsum) {
  size_t i = (size_t)blockIdx.x * blockDim.x + threadIdx.x;
  if (i >= (size_t)N_NODES * 256) return;
  int n = (int)(i >> 8);
  int h = ((int)i & 255) >> 6;
  out[i] /= rowsum[(size_t)h * N_NODES + n];
}

extern "C" void kernel_launch(void* const* d_in, const int* in_sizes, int n_in,
                              void* d_out, int out_size, void* d_ws, size_t ws_size,
                              hipStream_t stream) {
  (void)in_sizes; (void)n_in; (void)out_size; (void)ws_size;
  const float* x    = (const float*)d_in[0];
  const int*   idx  = (const int*)d_in[1];
  const float* elem = (const float*)d_in[2];
  const float* W1   = (const float*)d_in[3];
  const float* b1   = (const float*)d_in[4];
  const float* W2   = (const float*)d_in[5];
  const float* b2   = (const float*)d_in[6];
  const float* aw   = (const float*)d_in[7];
  const float* ab   = (const float*)d_in[8];
  float* out = (float*)d_out;

  char* ws = (char*)d_ws;
  size_t off = 0;
  auto alloc = [&](size_t bytes) -> void* {
    void* p = ws + off;
    off += (bytes + 255) & ~(size_t)255;
    return p;
  };
  f16*      xh     = (f16*)alloc((size_t)N_NODES * IN_DIM * sizeof(f16));
  f16*      w1t    = (f16*)alloc((size_t)N_HEADS * HID_DIM * IN_DIM * sizeof(f16));
  f16*      w2t    = (f16*)alloc((size_t)N_HEADS * OUT_DIM * HID_DIM * sizeof(f16));
  f16*      hid    = (f16*)alloc((size_t)N_NODES * HID_DIM * sizeof(f16));
  float*    feats  = (float*)alloc((size_t)N_NODES * OUT_DIM * sizeof(float));
  float*    score  = (float*)alloc((size_t)N_EDGES * sizeof(float));
  float*    rowsum = (float*)alloc((size_t)N_HEADS * N_NODES * sizeof(float));
  unsigned* maxu   = (unsigned*)alloc(N_HEADS * sizeof(unsigned));

  k_init<<<2048, 256, 0, stream>>>(out, rowsum, maxu);
  k_cvt_x<<<2048, 256, 0, stream>>>(x, xh);
  k_cvt_w1<<<(N_HEADS * HID_DIM * IN_DIM + 255) / 256, 256, 0, stream>>>(W1, w1t);
  k_cvt_w2<<<(N_HEADS * OUT_DIM * HID_DIM + 255) / 256, 256, 0, stream>>>(W2, w2t);

  for (int h = 0; h < N_HEADS; ++h) {
    k_gemm1<<<dim3(N_NODES / 16, HID_DIM / 64), 128, 0, stream>>>(xh, w1t, b1, hid, h);
    k_gemm2<<<dim3(N_NODES / 16), 128, 0, stream>>>(hid, w2t, b2, feats, h);
    k_score<<<(N_EDGES + 255) / 256, 256, 0, stream>>>(feats, idx, elem, aw, ab,
                                                       score, maxu, h);
    k_scatter<<<(N_EDGES * 32 + 255) / 256, 256, 0, stream>>>(feats, idx, score,
                                                              maxu, out, rowsum, h);
  }
  k_div<<<((size_t)N_NODES * 256 + 255) / 256, 256, 0, stream>>>(out, rowsum);
}
